// ImprovedVulnerabilityGNN_47021301957465
// MI455X (gfx1250) — compile-verified, matrix operand order checked by reading, not run on hardware
//
#include <hip/hip_runtime.h>
#include <hip/hip_bf16.h>
#include <math.h>

typedef __attribute__((ext_vector_type(16))) _Float16 v16h;
typedef __attribute__((ext_vector_type(8)))  float    v8f;

// ---------- helpers: order-preserving float<->uint for atomic max ----------
__device__ __forceinline__ unsigned f2ord(float f) {
  unsigned u = __float_as_uint(f);
  return (u & 0x80000000u) ? ~u : (u | 0x80000000u);
}
__device__ __forceinline__ float ord2f(unsigned u) {
  return (u & 0x80000000u) ? __uint_as_float(u & 0x7fffffffu)
                           : __uint_as_float(~u);
}
#define ORD_NEG_INF 0x007FFFFFu          // f2ord(-inf)
#define BN_INV_STD 0.99999500003749981f  // 1/sqrt(1+1e-5)

// ---- CDNA5 async global->LDS copy (16B), tracked by ASYNCcnt ----
__device__ __forceinline__ void async_cp16(const _Float16* gptr, _Float16* lptr) {
  unsigned lds_off = (unsigned)(size_t)lptr;           // low 32 bits = LDS offset
  unsigned long long ga = (unsigned long long)(size_t)gptr;
  asm volatile("global_load_async_to_lds_b128 %0, %1, off"
               :: "v"(lds_off), "v"(ga) : "memory");
}
__device__ __forceinline__ void wait_async0() {
  asm volatile("s_wait_asynccnt 0x0" ::: "memory");
}

// ---------------------------- fill kernel ----------------------------
__global__ void fill_u32(unsigned* __restrict__ p, long long n, unsigned v) {
  long long i = blockIdx.x * (long long)blockDim.x + threadIdx.x;
  if (i < n) p[i] = v;
}

// ------------- f32 -> f16 converters (one-time / per-layer prep) -------------
__global__ void cvt_f16(const float* __restrict__ in, _Float16* __restrict__ out,
                        long long n) {
  long long i = blockIdx.x * (long long)blockDim.x + threadIdx.x;
  if (i < n) out[i] = (_Float16)in[i];
}
// out[m*Kp+k] = k<Kin ? in[m*Kin+k] : 0   (row pad + convert)
__global__ void pad_cvt_rows(const float* __restrict__ in,
                             _Float16* __restrict__ out,
                             int M, int Kin, int Kp) {
  long long t = blockIdx.x * (long long)blockDim.x + threadIdx.x;
  if (t >= (long long)M * Kp) return;
  int k = (int)(t % Kp);
  long long m = t / Kp;
  out[t] = (k < Kin) ? (_Float16)in[m * Kin + k] : (_Float16)0.0f;
}
// WT[n*Kp+k] = k<K ? W[k*Nn+n] : 0   (transpose + pad + convert, one-time)
__global__ void transpose_cvt(const float* __restrict__ W,
                              _Float16* __restrict__ WT,
                              int K, int Nn, int Kp) {
  long long t = blockIdx.x * (long long)blockDim.x + threadIdx.x;
  if (t >= (long long)Nn * Kp) return;
  int k = (int)(t % Kp);
  long long n = t / Kp;
  WT[t] = (k < K) ? (_Float16)W[(long long)k * Nn + n] : (_Float16)0.0f;
}

// ------------------- WMMA GEMM: C[M,N] = A[M,Kp] @ BT[N,Kp]^T -------------------
// A: f16 row-major [M,Kp]; BT: f16 row-major [Nn,Kp] (pre-transposed weights);
// C: f32. Kp must be a multiple of 32. mode: 0=plain, 1=bias+relu.
#define BM 128
#define BNT 64
#define BK 32
#define LDAS 40  // halves per LDS row (80B): 16B-aligned chunks, conflict-free pairs

__global__ __launch_bounds__(256) void gemm_wmma(
    const _Float16* __restrict__ A, const _Float16* __restrict__ BT,
    const float* __restrict__ bias, float* __restrict__ C,
    int M, int Kp, int Nn, int mode) {
  __shared__ _Float16 As[BM * LDAS];
  __shared__ _Float16 Bs[BNT * LDAS];
  const int tid  = threadIdx.x;
  const int lane = tid & 31;
  const int wv   = tid >> 5;
  const int m0   = blockIdx.y * BM;
  const int n0   = blockIdx.x * BNT;

  v8f accs[4] = {v8f{}, v8f{}, v8f{}, v8f{}};

  const int rowA = wv * 16 + (lane & 15);
  const int kb   = (lane >> 4) * 8;  // 0 for lanes 0-15, 8 for lanes 16-31
  const int colL = lane & 15;

  // per-thread staging coordinates (16B chunks of 8 halves)
  const int ar0 = tid >> 1;                      // j=0: rows 0..127 (e=2*? no)
  (void)ar0;

  for (int k0 = 0; k0 < Kp; k0 += BK) {
    // ---- async stage A tile [128 x 32] f16: 2 x b128 per thread ----
#pragma unroll
    for (int j = 0; j < 2; ++j) {
      int e  = (j << 8) + tid;        // 0..511
      int r  = e >> 2;                // 0..127
      int c8 = (e & 3) << 3;          // 0,8,16,24
      int gm = m0 + r;
      gm = gm < M ? gm : (M - 1);     // branchless clamp (excess rows unstored)
      async_cp16(A + (long long)gm * Kp + (k0 + c8), &As[r * LDAS + c8]);
    }
    // ---- async stage B tile [64 x 32] f16 from BT: 1 x b128 per thread ----
    {
      int nn = tid >> 2;              // 0..63
      int c8 = (tid & 3) << 3;
      async_cp16(BT + (long long)(n0 + nn) * Kp + (k0 + c8), &Bs[nn * LDAS + c8]);
    }
    // prefetch next K tile into L2 while this one lands
    if (k0 + BK < Kp) {
      int pr = m0 + (tid >> 1);
      pr = pr < M ? pr : (M - 1);
      __builtin_prefetch(A + (long long)pr * Kp + (k0 + BK), 0, 1);
      __builtin_prefetch(BT + (long long)(n0 + (tid >> 2)) * Kp + (k0 + BK), 0, 1);
    }
    wait_async0();
    __syncthreads();

    // A fragment: lane holds row (lane&15); K pairs {kb..kb+7} in VGPRs 0-3,
    // {16+kb..16+kb+7} in VGPRs 4-7 (ISA 16-bit A-matrix 16x32 table).
    v16h a;
#pragma unroll
    for (int v = 0; v < 8; ++v) {
      int k = ((v & 4) << 2) + kb + ((v & 3) << 1);
      a[2 * v]     = As[rowA * LDAS + k];
      a[2 * v + 1] = As[rowA * LDAS + k + 1];
    }
    v16h b0, b1, b2, b3;
#pragma unroll
    for (int v = 0; v < 8; ++v) {
      int k = ((v & 4) << 2) + kb + ((v & 3) << 1);
      b0[2 * v] = Bs[(colL)      * LDAS + k]; b0[2 * v + 1] = Bs[(colL)      * LDAS + k + 1];
      b1[2 * v] = Bs[(16 + colL) * LDAS + k]; b1[2 * v + 1] = Bs[(16 + colL) * LDAS + k + 1];
      b2[2 * v] = Bs[(32 + colL) * LDAS + k]; b2[2 * v + 1] = Bs[(32 + colL) * LDAS + k + 1];
      b3[2 * v] = Bs[(48 + colL) * LDAS + k]; b3[2 * v + 1] = Bs[(48 + colL) * LDAS + k + 1];
    }
    accs[0] = __builtin_amdgcn_wmma_f32_16x16x32_f16(false, a, false, b0, (short)0, accs[0], false, false);
    accs[1] = __builtin_amdgcn_wmma_f32_16x16x32_f16(false, a, false, b1, (short)0, accs[1], false, false);
    accs[2] = __builtin_amdgcn_wmma_f32_16x16x32_f16(false, a, false, b2, (short)0, accs[2], false, false);
    accs[3] = __builtin_amdgcn_wmma_f32_16x16x32_f16(false, a, false, b3, (short)0, accs[3], false, false);
    __syncthreads();
  }

  // D layout: VGPR j -> row j + 8*(lane>=16), col = lane&15
  const int mbase = m0 + wv * 16 + ((lane >> 4) << 3);
#pragma unroll
  for (int j = 0; j < 8; ++j) {
    int gm = mbase + j;
    if (gm < M) {
#pragma unroll
      for (int t = 0; t < 4; ++t) {
        int col = n0 + t * 16 + colL;
        float v = accs[t][j];
        if (mode == 1) v = fmaxf(v + bias[col], 0.0f);
        C[(long long)gm * Nn + col] = v;
      }
    }
  }
}

// ------------- alpha_s / alpha_d: per (node, head) dot over C -------------
__global__ void alpha_kernel(const float* __restrict__ hW,
                             const float* __restrict__ a_s,
                             const float* __restrict__ a_d,
                             float* __restrict__ out_s,
                             float* __restrict__ out_d,
                             int Nnodes, int H, int C) {
  int wid  = blockIdx.x * (blockDim.x >> 5) + (threadIdx.x >> 5);
  int lane = threadIdx.x & 31;
  if (wid >= Nnodes * H) return;
  int n = wid / H, h = wid - n * H;
  const float* hp  = hW + (long long)n * H * C + (long long)h * C;
  const float* asp = a_s + h * C;
  const float* adp = a_d + h * C;
  float ps = 0.f, pd = 0.f;
  for (int c = lane; c < C; c += 32) {
    float v = hp[c];
    ps += v * asp[c];
    pd += v * adp[c];
  }
  for (int off = 16; off; off >>= 1) {
    ps += __shfl_xor(ps, off, 32);
    pd += __shfl_xor(pd, off, 32);
  }
  if (lane == 0) { out_s[wid] = ps; out_d[wid] = pd; }
}

// ----------------- edge pass 1: segment max of attention logit -----------------
__global__ void edge_max_kernel(const int* __restrict__ ei, int E, int Nnodes,
                                const float* __restrict__ as_,
                                const float* __restrict__ ad_,
                                unsigned* __restrict__ m, int H) {
  long long t = blockIdx.x * (long long)blockDim.x + threadIdx.x;
  long long tot = (long long)(E + Nnodes) * H;
  if (t >= tot) return;
  int h = (int)(t % H);
  long long e = t / H;
  int s, d;
  if (e < E) { s = ei[e]; d = ei[E + e]; } else { s = d = (int)(e - E); }
  float v = as_[(long long)s * H + h] + ad_[(long long)d * H + h];
  v = v > 0.f ? v : 0.2f * v;  // leaky_relu(0.2)
  atomicMax(&m[(long long)d * H + h], f2ord(v));
}

// --------- edge pass 2: w = exp(e - m[dst]); scatter w and w*h[src] ---------
__global__ void edge_accum_kernel(const int* __restrict__ ei, int E, int Nnodes,
                                  const float* __restrict__ as_,
                                  const float* __restrict__ ad_,
                                  const unsigned* __restrict__ m,
                                  const float* __restrict__ hW,
                                  float* __restrict__ ssum,
                                  float* __restrict__ acc, int H, int C) {
  long long wid = blockIdx.x * (long long)(blockDim.x >> 5) + (threadIdx.x >> 5);
  int lane = threadIdx.x & 31;
  long long tot = (long long)(E + Nnodes) * H;
  if (wid >= tot) return;
  int h = (int)(wid % H);
  long long e = wid / H;
  int s, d;
  if (e < E) { s = ei[e]; d = ei[E + e]; } else { s = d = (int)(e - E); }
  float v = as_[(long long)s * H + h] + ad_[(long long)d * H + h];
  v = v > 0.f ? v : 0.2f * v;
  float w = __expf(v - ord2f(m[(long long)d * H + h]));
  if (lane == 0) atomicAdd(&ssum[(long long)d * H + h], w);
  const float* hp = hW + (long long)s * H * C + (long long)h * C;
  float* ap = acc + (long long)d * H * C + (long long)h * C;
  for (int c = lane; c < C; c += 32) atomicAdd(&ap[c], w * hp[c]);
}

// ---- epilogue (concat): normalize + bias + BN + relu, writes f16 for next GEMM ----
__global__ void gat_epi_concat(const float* __restrict__ acc,
                               const float* __restrict__ ssum,
                               const float* __restrict__ bias,
                               const float* __restrict__ gamma,
                               const float* __restrict__ beta,
                               _Float16* __restrict__ out,
                               int Nnodes, int H, int C) {
  long long t = blockIdx.x * (long long)blockDim.x + threadIdx.x;
  long long tot = (long long)Nnodes * H * C;
  if (t >= tot) return;
  int HC = H * C;
  int f = (int)(t % HC);
  long long n = t / HC;
  int h = f / C;
  float v = acc[t] / (ssum[n * H + h] + 1e-16f) + bias[f];
  v = v * (gamma[f] * BN_INV_STD) + beta[f];
  out[t] = (_Float16)fmaxf(v, 0.f);
}

// ---------- epilogue (mean heads): mean_h + bias + BN + relu -> f32 ----------
__global__ void gat_epi_mean(const float* __restrict__ acc,
                             const float* __restrict__ ssum,
                             const float* __restrict__ bias,
                             const float* __restrict__ gamma,
                             const float* __restrict__ beta,
                             float* __restrict__ out,
                             int Nnodes, int H, int C) {
  long long t = blockIdx.x * (long long)blockDim.x + threadIdx.x;
  long long tot = (long long)Nnodes * C;
  if (t >= tot) return;
  int c = (int)(t % C);
  long long n = t / C;
  float v = 0.f;
  for (int h = 0; h < H; ++h)
    v += acc[(n * H + h) * (long long)C + c] / (ssum[n * H + h] + 1e-16f);
  v = v / (float)H + bias[c];
  v = v * (gamma[c] * BN_INV_STD) + beta[c];
  out[n * C + c] = fmaxf(v, 0.f);
}

// ---------------------------- graph pooling ----------------------------
__global__ void pool_kernel(const float* __restrict__ x,
                            const int* __restrict__ batch,
                            float* __restrict__ addp,
                            unsigned* __restrict__ maxp,
                            float* __restrict__ cnt, int Nnodes, int C) {
  long long t = blockIdx.x * (long long)blockDim.x + threadIdx.x;
  long long tot = (long long)Nnodes * C;
  if (t >= tot) return;
  int c = (int)(t % C);
  int n = (int)(t / C);
  int g = batch[n];
  float v = x[t];
  atomicAdd(&addp[(long long)g * C + c], v);
  atomicMax(&maxp[(long long)g * C + c], f2ord(v));
  if (c == 0) atomicAdd(&cnt[g], 1.0f);
}

__global__ void pool_fin(const float* __restrict__ addp,
                         const unsigned* __restrict__ maxp,
                         const float* __restrict__ cnt,
                         float* __restrict__ pooled, int G, int C) {
  int t = blockIdx.x * blockDim.x + threadIdx.x;
  if (t >= G * C) return;
  int c = t % C, g = t / C;
  float a = addp[t];
  float mx = ord2f(maxp[t]);
  if (!isfinite(mx)) mx = 0.f;
  float mean = a / fmaxf(cnt[g], 1.0f);
  pooled[(long long)g * 3 * C + c]         = mean;
  pooled[(long long)g * 3 * C + C + c]     = mx;
  pooled[(long long)g * 3 * C + 2 * C + c] = a;
}

// ---------- small dense: out[G,Nout] = act(A[G,K] @ W[K,Nout] + b) ----------
__global__ void dense_kernel(const float* __restrict__ A,
                             const float* __restrict__ W,
                             const float* __restrict__ b,
                             float* __restrict__ out,
                             int G, int K, int Nout, int act) {
  int t = blockIdx.x * blockDim.x + threadIdx.x;
  if (t >= G * Nout) return;
  int n = t % Nout, g = t / Nout;
  float s = b[n];
  const float* ar = A + (long long)g * K;
  for (int k = 0; k < K; ++k) s += ar[k] * W[(long long)k * Nout + n];
  if (act == 1) s = fmaxf(s, 0.f);
  else if (act == 2) s = 1.f / (1.f + __expf(-s));
  out[t] = s;
}

// =============================== launcher ===============================
extern "C" void kernel_launch(void* const* d_in, const int* in_sizes, int n_in,
                              void* d_out, int out_size, void* d_ws, size_t ws_size,
                              hipStream_t stream) {
  (void)n_in; (void)out_size; (void)ws_size;
  const float* x     = (const float*)d_in[0];
  const int*   ei    = (const int*)d_in[1];
  const int*   batch = (const int*)d_in[2];
  const float* Win   = (const float*)d_in[3];
  const float* b_in  = (const float*)d_in[4];
  const float* W1  = (const float*)d_in[5],  *as1 = (const float*)d_in[6],  *ad1 = (const float*)d_in[7];
  const float* b1  = (const float*)d_in[8],  *g1  = (const float*)d_in[9],  *be1 = (const float*)d_in[10];
  const float* W2  = (const float*)d_in[11], *as2 = (const float*)d_in[12], *ad2 = (const float*)d_in[13];
  const float* b2  = (const float*)d_in[14], *g2  = (const float*)d_in[15], *be2 = (const float*)d_in[16];
  const float* W3  = (const float*)d_in[17], *as3 = (const float*)d_in[18], *ad3 = (const float*)d_in[19];
  const float* b3  = (const float*)d_in[20], *g3  = (const float*)d_in[21], *be3 = (const float*)d_in[22];
  const float* Wp  = (const float*)d_in[23], *bp  = (const float*)d_in[24];
  const float* Wv1 = (const float*)d_in[25], *bv1 = (const float*)d_in[26];
  const float* Wv2 = (const float*)d_in[27], *bv2 = (const float*)d_in[28];
  const float* Wv3 = (const float*)d_in[29], *bv3 = (const float*)d_in[30];
  const float* Wt1 = (const float*)d_in[31], *bt1 = (const float*)d_in[32];
  const float* Wt2 = (const float*)d_in[33], *bt2 = (const float*)d_in[34];
  const float* Wt3 = (const float*)d_in[35], *bt3 = (const float*)d_in[36];
  const float* Wc1 = (const float*)d_in[37], *bc1 = (const float*)d_in[38];
  const float* Wc2 = (const float*)d_in[39], *bc2 = (const float*)d_in[40];

  const int N = in_sizes[2];      // 50000 nodes
  const int E = in_sizes[1] / 2;  // 200000 edges (self-loops added on the fly)
  const int G = 64;
  float* dout = (float*)d_out;

  // ---- workspace partition (byte-based, 256B aligned) ----
  char* wsb = (char*)d_ws;
  size_t off = 0;
  auto allocB = [&](size_t bytes) {
    void* p = wsb + off;
    off += (bytes + 255) & ~(size_t)255;
    return p;
  };
  float*    h0   = (float*)allocB((size_t)N * 256 * 4);     // f32 scratch / x3
  float*    B1g  = (float*)allocB((size_t)N * 1024 * 4);    // hW (GEMM out)
  float*    B2g  = (float*)allocB((size_t)N * 1024 * 4);    // attention accumulator
  float*    alS  = (float*)allocB((size_t)N * 8 * 4);
  float*    alD  = (float*)allocB((size_t)N * 8 * 4);
  unsigned* mbuf = (unsigned*)allocB((size_t)N * 8 * 4);
  float*    sbuf = (float*)allocB((size_t)N * 8 * 4);
  float*    addp = (float*)allocB((size_t)G * 256 * 4);
  unsigned* maxp = (unsigned*)allocB((size_t)G * 256 * 4);
  float*    cnt  = (float*)allocB((size_t)G * 4);
  float*    pld  = (float*)allocB((size_t)G * 768 * 4);
  float*    gv   = (float*)allocB((size_t)G * 256 * 4);
  float*    t1   = (float*)allocB((size_t)G * 128 * 4);
  float*    t2   = (float*)allocB((size_t)G * 64 * 4);
  _Float16* xh   = (_Float16*)allocB((size_t)N * 64 * 2);   // x padded 62->64, f16
  _Float16* h16  = (_Float16*)allocB((size_t)N * 256 * 2);  // input-proj acts f16
  _Float16* A16  = (_Float16*)allocB((size_t)N * 1024 * 2); // GAT acts f16
  _Float16* WinT = (_Float16*)allocB((size_t)256 * 64 * 2);
  _Float16* W1T  = (_Float16*)allocB((size_t)1024 * 256 * 2);
  _Float16* W2T  = (_Float16*)allocB((size_t)1024 * 1024 * 2);
  _Float16* W3T  = (_Float16*)allocB((size_t)1024 * 1024 * 2);

  auto fill = [&](void* p, long long n, unsigned v) {
    fill_u32<<<(unsigned)((n + 255) / 256), 256, 0, stream>>>((unsigned*)p, n, v);
  };
  auto gemm = [&](const _Float16* A, const _Float16* BT, const float* bias,
                  float* Cm, int M, int Kp, int Nn, int mode) {
    dim3 grid(Nn / BNT, (M + BM - 1) / BM);
    gemm_wmma<<<grid, 256, 0, stream>>>(A, BT, bias, Cm, M, Kp, Nn, mode);
  };
  auto run_gat = [&](const _Float16* Xin, int Kp, const _Float16* WT,
                     const float* a_s, const float* a_d, int H, int C) {
    gemm(Xin, WT, nullptr, B1g, N, Kp, H * C, 0);
    int waves = N * H;
    alpha_kernel<<<(waves + 7) / 8, 256, 0, stream>>>(B1g, a_s, a_d, alS, alD, N, H, C);
    fill(mbuf, (long long)N * H, ORD_NEG_INF);
    fill(sbuf, (long long)N * H, 0u);
    fill(B2g, (long long)N * H * C, 0u);
    long long tot = (long long)(E + N) * H;
    edge_max_kernel<<<(unsigned)((tot + 255) / 256), 256, 0, stream>>>(
        ei, E, N, alS, alD, mbuf, H);
    edge_accum_kernel<<<(unsigned)((tot + 7) / 8), 256, 0, stream>>>(
        ei, E, N, alS, alD, mbuf, B1g, sbuf, B2g, H, C);
  };

  // ---- one-time prep: f16 inputs + transposed f16 weights ----
  pad_cvt_rows<<<(unsigned)(((long long)N * 64 + 255) / 256), 256, 0, stream>>>(
      x, xh, N, 62, 64);
  transpose_cvt<<<(256 * 64 + 255) / 256, 256, 0, stream>>>(Win, WinT, 62, 256, 64);
  transpose_cvt<<<(1024 * 256 + 255) / 256, 256, 0, stream>>>(W1, W1T, 256, 1024, 256);
  transpose_cvt<<<(1024 * 1024 + 255) / 256, 256, 0, stream>>>(W2, W2T, 1024, 1024, 1024);
  transpose_cvt<<<(1024 * 1024 + 255) / 256, 256, 0, stream>>>(W3, W3T, 1024, 1024, 1024);

  // ---- input projection: h0 = relu(x @ Win + b_in); keep f16 copy ----
  gemm(xh, WinT, b_in, h0, N, 64, 256, 1);
  cvt_f16<<<(unsigned)(((long long)N * 256 + 255) / 256), 256, 0, stream>>>(
      h0, h16, (long long)N * 256);

  // ---- GAT layer 1 (256 -> 8x128, concat) ----
  run_gat(h16, 256, W1T, as1, ad1, 8, 128);
  {
    long long tot = (long long)N * 1024;
    gat_epi_concat<<<(unsigned)((tot + 255) / 256), 256, 0, stream>>>(
        B2g, sbuf, b1, g1, be1, A16, N, 8, 128);
  }
  // ---- GAT layer 2 (1024 -> 8x128, concat) ----
  run_gat(A16, 1024, W2T, as2, ad2, 8, 128);
  {
    long long tot = (long long)N * 1024;
    gat_epi_concat<<<(unsigned)((tot + 255) / 256), 256, 0, stream>>>(
        B2g, sbuf, b2, g2, be2, A16, N, 8, 128);
  }
  // ---- GAT layer 3 (1024 -> 4x256, mean) -> h0 [N,256] f32 ----
  run_gat(A16, 1024, W3T, as3, ad3, 4, 256);
  {
    long long tot = (long long)N * 256;
    gat_epi_mean<<<(unsigned)((tot + 255) / 256), 256, 0, stream>>>(
        B2g, sbuf, b3, g3, be3, h0, N, 4, 256);
  }

  // ---- pooling ----
  fill(addp, (long long)G * 256, 0u);
  fill(maxp, (long long)G * 256, ORD_NEG_INF);
  fill(cnt, G, 0u);
  {
    long long tot = (long long)N * 256;
    pool_kernel<<<(unsigned)((tot + 255) / 256), 256, 0, stream>>>(
        h0, batch, addp, maxp, cnt, N, 256);
  }
  pool_fin<<<(G * 256 + 255) / 256, 256, 0, stream>>>(addp, maxp, cnt, pld, G, 256);

  // ---- heads ----
  auto dense = [&](const float* A, const float* W, const float* b, float* o,
                   int K, int Nout, int act) {
    dense_kernel<<<(G * Nout + 255) / 256, 256, 0, stream>>>(A, W, b, o, G, K, Nout, act);
  };
  dense(pld, Wp, bp, gv, 768, 256, 1);         // g = relu(pooled@Wp+bp)
  dense(gv, Wv1, bv1, t1, 256, 128, 1);        // vuln head
  dense(t1, Wv2, bv2, t2, 128, 64, 1);
  dense(t2, Wv3, bv3, dout, 64, 2, 0);         // -> d_out[0:128)
  dense(gv, Wt1, bt1, t1, 256, 128, 1);        // vtype head
  dense(t1, Wt2, bt2, t2, 128, 64, 1);
  dense(t2, Wt3, bt3, dout + 128, 64, 10, 0);  // -> d_out[128:768)
  dense(gv, Wc1, bc1, t2, 256, 64, 1);         // conf head
  dense(t2, Wc2, bc2, dout + 768, 64, 1, 2);   // -> d_out[768:832)
}
